// GlimpseSensor_66228395705152
// MI455X (gfx1250) — matrix operand order repaired and checked by reading, not compiled, not run on hardware
//
#include <hip/hip_runtime.h>
#include <math.h>

// ---------------------------------------------------------------------------
// GlimpseSensor on MI455X (gfx1250): bf16 WMMA implicit-GEMM convs + K-split
// WMMA GEMM (LDS-staged activations) for the big linear layers.
// ---------------------------------------------------------------------------

typedef __attribute__((ext_vector_type(16))) __bf16   bf16x16;
typedef __attribute__((ext_vector_type(8)))  __bf16   bf16x8;
typedef __attribute__((ext_vector_type(4)))  __bf16   bf16x4;
typedef __attribute__((ext_vector_type(8)))  float    f32x8;
typedef __attribute__((ext_vector_type(4)))  float    f32x4;
typedef __attribute__((ext_vector_type(4)))  unsigned u32x4;

#define BATCH 128
#define IMGSZ 224
#define NOUT  768          // SENSOR*SENSOR*C_IN = 16*16*3

__device__ __forceinline__ __bf16 f2bf(float f) { return (__bf16)f; }

// K index held by element j of a 16-bit A-fragment (16x32), lane-half h.
__device__ __forceinline__ int akidx(int h, int j) {
  int v = j >> 1, p = j & 1;
  return 16 * h + 2 * v + p;
}

__device__ __forceinline__ bf16x16 cat16(bf16x8 lo, bf16x8 hi) {
  return __builtin_shufflevector(lo, hi, 0, 1, 2, 3, 4, 5, 6, 7,
                                 8, 9, 10, 11, 12, 13, 14, 15);
}

// ---------------------------------------------------------------------------
// Crop extraction: padded image -> bf16 crops, pixel-major [b][pix][cin].
// ---------------------------------------------------------------------------
__global__ void crop_kernel(const float* __restrict__ imgs,
                            const float* __restrict__ locs,
                            __bf16* __restrict__ crop) {
  const int b = blockIdx.x;
  const float lx = locs[2 * b], ly = locs[2 * b + 1];
  const int cH = 32 + (int)rintf((lx + 1.f) * 0.5f * 224.f);
  const int cW = 32 + (int)rintf((ly + 1.f) * 0.5f * 224.f);
  const int    gs[3]  = {16, 32, 64};
  const size_t off[3] = {0, 98304, 491520};   // element offsets
  for (int i = 0; i < 3; ++i) {
    const int g = gs[i], tot = 3 * g * g;
    __bf16* dst = crop + off[i] + (size_t)b * tot;
    const int oy = cH - g / 2 - 32, ox = cW - g / 2 - 32;
    for (int t = threadIdx.x; t < tot; t += blockDim.x) {
      int ci = t % 3, pix = t / 3;
      int y = pix / g, x = pix % g;
      int iy = oy + y, ix = ox + x;
      float v = 0.f;
      if (iy >= 0 && iy < IMGSZ && ix >= 0 && ix < IMGSZ)
        v = imgs[((size_t)b * 3 + ci) * (IMGSZ * IMGSZ) + (size_t)iy * IMGSZ + ix];
      dst[t] = f2bf(v);
    }
  }
}

__global__ void zero_kernel(float* __restrict__ p, int n) {
  int i = blockIdx.x * blockDim.x + threadIdx.x;
  if (i < n) p[i] = 0.f;
}

// ---------------------------------------------------------------------------
// Pre-pack conv weights into WMMA A-fragment order (bf16):
//   pk[(tap*32 + lane)*32 + j]      = A0 element j (couts 0..15)
//   pk[(tap*32 + lane)*32 + 16 + j] = A1 element j (couts 16..31)
// ---------------------------------------------------------------------------
__global__ void pack_conv_w(const float* __restrict__ w,
                            __bf16* __restrict__ pk, int first) {
  const int lane = threadIdx.x & 31;
  const int tap = blockIdx.x;
  const int m = lane & 15, h = lane >> 4;
  __bf16* dst = pk + ((size_t)tap * 32 + lane) * 32;
#pragma unroll
  for (int j = 0; j < 16; ++j) {
    int K = akidx(h, j);
    float a0, a1;
    if (first) {                        // K = cin*9 + tap, 27 real, zero-pad
      a0 = (K < 27) ? w[m * 27 + K] : 0.f;
      a1 = (K < 27) ? w[(m + 16) * 27 + K] : 0.f;
    } else {                            // K = cin within tap, w[32][32][9]
      a0 = w[((size_t)m * 32 + K) * 9 + tap];
      a1 = w[((size_t)(m + 16) * 32 + K) * 9 + tap];
    }
    dst[j]      = f2bf(a0);
    dst[16 + j] = f2bf(a1);
  }
}

// ---------------------------------------------------------------------------
// 3x3 SAME conv + bias + ReLU as implicit GEMM with V_WMMA_F32_16X16X32_BF16.
// G templated: shifts + immediate offsets; py / sy bounds are wave-uniform
// (tiles are 16-pixel row segments), only row-edge lanes need the x clamp.
// ---------------------------------------------------------------------------
template <int G, bool FIRST, bool CHMAJOR>
__global__ void conv3x3_wmma(const __bf16* __restrict__ in,
                             const __bf16* __restrict__ pkw,  // packed A frags
                             const float* __restrict__ bias,
                             __bf16* __restrict__ out) {
  constexpr int GG = G * G;
  const int b    = blockIdx.x;
  const int lane = threadIdx.x & 31;
  const int n = lane & 15;        // N column (pixel)
  const int h = lane >> 4;        // lane half -> K split
  const int tile = __builtin_amdgcn_readfirstlane(blockIdx.y * 4 + (threadIdx.x >> 5));
  const int pixbase = tile * 16;
  const int py  = pixbase / G;    // scalar: tile is a row segment
  const int pxb = pixbase % G;    // scalar
  const int px  = pxb + n;
  const int pix = pixbase + n;

  const bf16x16* pw = reinterpret_cast<const bf16x16*>(pkw);  // [tap][lane][2]

  f32x8 c0 = {}; f32x8 c1 = {};

  if (FIRST) {
    bf16x16 a0 = pw[lane * 2], a1 = pw[lane * 2 + 1];
    bf16x16 bm;
#pragma unroll
    for (int j = 0; j < 16; ++j) {
      int K = 16 * h + j;                 // K = cin*9 + tap
      __bf16 v = f2bf(0.f);
      if (K < 27) {
        int ci = K / 9, tap = K % 9;
        int sy = py + tap / 3 - 1, sx = px + tap % 3 - 1;
        if (sy >= 0 && sy < G && sx >= 0 && sx < G)
          v = in[((size_t)b * GG + sy * G + sx) * 3 + ci];
      }
      bm[j] = v;
    }
    c0 = __builtin_amdgcn_wmma_f32_16x16x32_bf16(false, a0, false, bm, (short)0, c0, false, false);
    c1 = __builtin_amdgcn_wmma_f32_16x16x32_bf16(false, a1, false, bm, (short)0, c1, false, false);
  } else {
    const __bf16* base = in + (size_t)b * GG * 32 + 16 * h;
#pragma unroll
    for (int tap = 0; tap < 9; ++tap) {
      const int dy = tap / 3 - 1, dx = tap % 3 - 1;
      const int sy = py + dy;                    // scalar
      bf16x16 bm = {};
      if (sy >= 0 && sy < G) {                   // s_cbranch (uniform)
        const int sx  = px + dx;
        const int sxc = sx < 0 ? 0 : (sx > G - 1 ? G - 1 : sx);
        bm = *reinterpret_cast<const bf16x16*>(base + (size_t)(sy * G + sxc) * 32);
        if (sx != sxc) bm = (bf16x16){};         // zero edge lane (cndmask)
      }
      bf16x16 a0 = pw[(tap * 32 + lane) * 2];
      bf16x16 a1 = pw[(tap * 32 + lane) * 2 + 1];
      c0 = __builtin_amdgcn_wmma_f32_16x16x32_bf16(false, a0, false, bm, (short)0, c0, false, false);
      c1 = __builtin_amdgcn_wmma_f32_16x16x32_bf16(false, a1, false, bm, (short)0, c1, false, false);
    }
  }

  // Epilogue: C/D layout row M = r + 8h, column N = lane&15.
  if (!CHMAJOR) {
    bf16x8 p0, p1;
#pragma unroll
    for (int r = 0; r < 8; ++r) {
      int co0 = 8 * h + r, co1 = co0 + 16;
      p0[r] = f2bf(fmaxf(c0[r] + bias[co0], 0.f));
      p1[r] = f2bf(fmaxf(c1[r] + bias[co1], 0.f));
    }
    *reinterpret_cast<bf16x8*>(out + ((size_t)b * GG + pix) * 32 + 8 * h)      = p0;
    *reinterpret_cast<bf16x8*>(out + ((size_t)b * GG + pix) * 32 + 16 + 8 * h) = p1;
  } else {
#pragma unroll
    for (int r = 0; r < 8; ++r) {
      int co0 = 8 * h + r, co1 = co0 + 16;
      out[((size_t)b * 32 + co0) * GG + pix] = f2bf(fmaxf(c0[r] + bias[co0], 0.f));
      out[((size_t)b * 32 + co1) * GG + pix] = f2bf(fmaxf(c1[r] + bias[co1], 0.f));
    }
  }
}

// ---------------------------------------------------------------------------
// Linear: partial[b][n] += act[b][k] * W[n][k].  Block = 4 waves = 4 N-tiles
// sharing a double-buffered LDS stage of the 128-row x 32-K act chunk
// (80B row pitch to spread DS banks). Each wave covers all 128 batch rows
// (8 accumulators) so each weight is loaded exactly once; K-split slices are
// reduced with f32 atomics into a zeroed partial buffer.
// ---------------------------------------------------------------------------
template <int IN_DIM, int CPS>
__global__ void linear_wmma(const __bf16* __restrict__ act,  // [128][IN_DIM]
                            const float* __restrict__ w,     // [768][IN_DIM]
                            float* __restrict__ partial) {   // [128][768]
  constexpr int PITCH = 40;                    // bf16 per LDS row (80 B)
  __shared__ __bf16 smem[2][BATCH * PITCH];

  const int t    = threadIdx.x;                // 0..127 == batch row to stage
  const int lane = threadIdx.x & 31;
  const int n = lane & 15;
  const int h = lane >> 4;
  const int nt = __builtin_amdgcn_readfirstlane(blockIdx.x * 4 + (threadIdx.x >> 5));
  const int nbase  = nt * 16;
  const int kstart = blockIdx.y * CPS * 32;

  f32x8 acc[8];
#pragma unroll
  for (int mt = 0; mt < 8; ++mt) acc[mt] = (f32x8){};

  auto stage = [&](int c, int buf) {
    const u32x4* src =
        reinterpret_cast<const u32x4*>(act + (size_t)t * IN_DIM + kstart + c * 32);
    u32x4* dst = reinterpret_cast<u32x4*>(&smem[buf][t * PITCH]);
    dst[0] = src[0]; dst[1] = src[1]; dst[2] = src[2]; dst[3] = src[3];
  };

  stage(0, 0);
  __syncthreads();

  for (int c = 0; c < CPS; ++c) {
    if (c + 1 < CPS) stage(c + 1, (c + 1) & 1);

    // B fragment: 16 consecutive fp32 weights per lane -> 4x b128 + v_cvt
    const float* wp = w + (size_t)(nbase + n) * IN_DIM + kstart + c * 32 + 16 * h;
    __builtin_prefetch(wp + 64, 0, 1);          // global_prefetch_b8
    const f32x4* wp4 = reinterpret_cast<const f32x4*>(wp);
    f32x4 w0 = wp4[0], w1 = wp4[1], w2 = wp4[2], w3 = wp4[3];
    bf16x8 bl = __builtin_shufflevector(__builtin_convertvector(w0, bf16x4),
                                        __builtin_convertvector(w1, bf16x4),
                                        0, 1, 2, 3, 4, 5, 6, 7);
    bf16x8 bh = __builtin_shufflevector(__builtin_convertvector(w2, bf16x4),
                                        __builtin_convertvector(w3, bf16x4),
                                        0, 1, 2, 3, 4, 5, 6, 7);
    bf16x16 bm = cat16(bl, bh);

    const __bf16* sb = smem[c & 1];
#pragma unroll
    for (int mt = 0; mt < 8; ++mt) {
      // A fragment for lane-half h = dword quads [2h] and [2h+1] of the row.
      const u32x4* lp = reinterpret_cast<const u32x4*>(sb + (mt * 16 + n) * PITCH);
      bf16x8 lo = __builtin_bit_cast(bf16x8, lp[2 * h]);
      bf16x8 hi = __builtin_bit_cast(bf16x8, lp[2 * h + 1]);
      acc[mt] = __builtin_amdgcn_wmma_f32_16x16x32_bf16(false, cat16(lo, hi), false, bm,
                                                        (short)0, acc[mt], false, false);
    }
    __syncthreads();
  }

#pragma unroll
  for (int mt = 0; mt < 8; ++mt) {
#pragma unroll
    for (int r = 0; r < 8; ++r) {
      int brow = mt * 16 + 8 * h + r;
      atomicAdd(partial + (size_t)brow * NOUT + nbase + n, acc[mt][r]);
    }
  }
}

// ---------------------------------------------------------------------------
// bias + sigmoid + channel-concat reshape -> (B, 9, 16, 16)
// ---------------------------------------------------------------------------
__global__ void epilogue_kernel(const float* __restrict__ part,
                                const float* __restrict__ b0,
                                const float* __restrict__ b1,
                                const float* __restrict__ b2,
                                float* __restrict__ out, int total) {
  int idx = blockIdx.x * blockDim.x + threadIdx.x;
  if (idx >= total) return;
  int b   = idx / (9 * 256);
  int rem = idx % (9 * 256);
  int ch  = rem >> 8;
  int pos = rem & 255;
  int i = ch / 3, c = ch % 3;
  int j = c * 256 + pos;
  const float* bb = (i == 0) ? b0 : (i == 1) ? b1 : b2;
  float v = part[(size_t)i * BATCH * NOUT + (size_t)b * NOUT + j] + bb[j];
  out[idx] = 1.f / (1.f + expf(-v));
}

// ---------------------------------------------------------------------------
template <int G, int SLICES, int CPS>
static void run_glimpse(const __bf16* crop, const float* c1w, const float* c1b,
                        const float* c2w, const float* c2b,
                        const float* c3w, const float* c3b,
                        const float* lw, float* part,
                        __bf16* actA, __bf16* actB,
                        __bf16* pk1, __bf16* pk2, __bf16* pk3,
                        hipStream_t stream) {
  constexpr int GG = G * G;
  pack_conv_w<<<1, 32, 0, stream>>>(c1w, pk1, 1);
  pack_conv_w<<<9, 32, 0, stream>>>(c2w, pk2, 0);
  pack_conv_w<<<9, 32, 0, stream>>>(c3w, pk3, 0);

  dim3 cgrid(BATCH, GG / 64);  // (GG/16 pixel tiles) / 4 waves per block
  conv3x3_wmma<G, true,  false><<<cgrid, 128, 0, stream>>>(crop, pk1, c1b, actA);
  conv3x3_wmma<G, false, false><<<cgrid, 128, 0, stream>>>(actA, pk2, c2b, actB);
  conv3x3_wmma<G, false, true ><<<cgrid, 128, 0, stream>>>(actB, pk3, c3b, actA);

  linear_wmma<GG * 32, CPS><<<dim3(12, SLICES), 128, 0, stream>>>(actA, lw, part);
}

extern "C" void kernel_launch(void* const* d_in, const int* in_sizes, int n_in,
                              void* d_out, int out_size, void* d_ws, size_t ws_size,
                              hipStream_t stream) {
  const float* imgs = (const float*)d_in[0];
  const float* locs = (const float*)d_in[1];
  const float* c1w  = (const float*)d_in[2];
  const float* c1b  = (const float*)d_in[3];
  const float* c2w  = (const float*)d_in[4];
  const float* c2b  = (const float*)d_in[5];
  const float* c3w  = (const float*)d_in[6];
  const float* c3b  = (const float*)d_in[7];
  const float* lw[3] = {(const float*)d_in[8], (const float*)d_in[10], (const float*)d_in[12]};
  const float* lb[3] = {(const float*)d_in[9], (const float*)d_in[11], (const float*)d_in[13]};
  float* out = (float*)d_out;

  char* ws = (char*)d_ws;
  __bf16* crop = (__bf16*)(ws);                          // 4.13 MB (bf16 crops)
  __bf16* actA = (__bf16*)(ws + ((size_t)8  << 20));     // 33.6 MB (g=64 acts)
  __bf16* actB = (__bf16*)(ws + ((size_t)44 << 20));     // 33.6 MB
  float*  part = (float*) (ws + ((size_t)80 << 20));     // 3*128*768 f32
  __bf16* pk1  = (__bf16*)(ws + ((size_t)84 << 20));              // 2 KB
  __bf16* pk2  = (__bf16*)(ws + ((size_t)84 << 20) + (32 << 10)); // 18 KB
  __bf16* pk3  = (__bf16*)(ws + ((size_t)84 << 20) + (64 << 10)); // 18 KB

  crop_kernel<<<BATCH, 256, 0, stream>>>(imgs, locs, crop);
  zero_kernel<<<(3 * BATCH * NOUT) / 256, 256, 0, stream>>>(part, 3 * BATCH * NOUT);

  run_glimpse<16, 16, 16>(crop,          c1w + 0 * 32 * 27, c1b + 0,
                          c2w + 0,       c2b + 0, c3w + 0, c3b + 0,
                          lw[0], part + 0,
                          actA, actB, pk1, pk2, pk3, stream);
  run_glimpse<32, 64, 16>(crop + 98304,  c1w + 1 * 32 * 27, c1b + 32,
                          c2w + 1 * 32 * 32 * 9, c2b + 32,
                          c3w + 1 * 32 * 32 * 9, c3b + 32,
                          lw[1], part + (size_t)1 * BATCH * NOUT,
                          actA, actB, pk1, pk2, pk3, stream);
  run_glimpse<64, 128, 32>(crop + 491520, c1w + 2 * 32 * 27, c1b + 64,
                          c2w + 2 * 32 * 32 * 9, c2b + 64,
                          c3w + 2 * 32 * 32 * 9, c3b + 64,
                          lw[2], part + (size_t)2 * BATCH * NOUT,
                          actA, actB, pk1, pk2, pk3, stream);

  const int total = BATCH * 9 * 256;
  epilogue_kernel<<<(total + 255) / 256, 256, 0, stream>>>(part, lb[0], lb[1], lb[2],
                                                           out, total);
}
